// ForgetGRU_23390391894968
// MI455X (gfx1250) — matrix operand order repaired and checked by reading, not compiled, not run on hardware
//
#include <hip/hip_runtime.h>

#define BB 128
#define TT 128
#define II 512
#define HH 512
#define EE 256
#define LL 2
#define KK 512   // I == H == 512: uniform GEMM inner dimension
#define NBLK 32  // persistent grid: 32 blocks (block = nb slice), 8 waves (wave = mb slice)

// LDS weight tile: 12 matrices x 16 rows, each row padded to 520 halves so
// consecutive rows are 1040 B apart (1040 % 256 = 16 -> lane n of a b128
// fragment read starts at bank 4n: conflict-free across 64 banks).
#define KPAD 520
#define LDS_ROWS (LL * 6 * 16)                  // 192
#define LDS_W_ELEMS (LDS_ROWS * KPAD)           // 99840 halves
#define LDS_BYTES   (LDS_W_ELEMS * 2)           // 199680 B (<= 320KB WGP LDS)

typedef __attribute__((ext_vector_type(16))) __bf16          bf16x16;
typedef __attribute__((ext_vector_type(8)))  float           f32x8;
typedef __attribute__((ext_vector_type(8)))  unsigned short  u16x8;
typedef __attribute__((ext_vector_type(16))) unsigned short  u16x16;

// ---------------------------------------------------------------- helpers

__device__ __forceinline__ unsigned short f2bf(float f) {
  unsigned int u = __builtin_bit_cast(unsigned int, f);
  unsigned int r = u + 0x7FFFu + ((u >> 16) & 1u);   // round-to-nearest-even
  return (unsigned short)(r >> 16);
}

__device__ __forceinline__ float sigmoidf_(float x) {
  return 1.0f / (1.0f + expf(-x));
}

// A-fragment (16x32 bf16, M x K). Lane (m = lane&15) holds K = base..base+15
// with base = (lane>=16 ? 16 : 0). Two b128 loads.
__device__ __forceinline__ bf16x16 load_a_frag(const unsigned short* p) {
  u16x8 lo = *(const u16x8*)(p);
  u16x8 hi = *(const u16x8*)(p + 8);
  u16x16 v;
#pragma unroll
  for (int i = 0; i < 8; ++i) { v[i] = lo[i]; v[i + 8] = hi[i]; }
  return __builtin_bit_cast(bf16x16, v);
}

// B-fragment (32x16 bf16, K x N). Lane (n = lane&15) holds K = base..base+15,
// base = (lane>=16 ? 16 : 0). Contiguous in k -> two b128 loads (global or LDS).
__device__ __forceinline__ bf16x16 load_b_frag(const unsigned short* p) {
  u16x8 lo = *(const u16x8*)(p);
  u16x8 hi = *(const u16x8*)(p + 8);
  u16x16 v;
#pragma unroll
  for (int i = 0; i < 8; ++i) { v[i] = lo[i]; v[i + 8] = hi[i]; }
  return __builtin_bit_cast(bf16x16, v);
}

__device__ __forceinline__ f32x8 wmma_bf16(bf16x16 a, bf16x16 b, f32x8 c) {
  return __builtin_amdgcn_wmma_f32_16x16x32_bf16(
      false, a, false, b, (short)0, c, false, false);
}

// ---------------------------------------------------------------- small kernels

__global__ void cvt_f32_to_bf16(const float* __restrict__ in,
                                unsigned short* __restrict__ out, int n) {
  int i = blockIdx.x * blockDim.x + threadIdx.x;
  int stride = gridDim.x * blockDim.x;
  for (; i < n; i += stride) out[i] = f2bf(in[i]);
}

// er[l][b][h] = sum_e extra[b][e] * W_er[l][h][e]   (tiny, once)
__global__ void er_kernel(const float* __restrict__ extra,
                          const float* __restrict__ W_er,
                          float* __restrict__ er) {
  int idx = blockIdx.x * blockDim.x + threadIdx.x;
  if (idx >= LL * BB * HH) return;
  int l = idx / (BB * HH);
  int rem = idx % (BB * HH);
  int b = rem / HH, h = rem % HH;
  const float* xr = extra + b * EE;
  const float* wr = W_er + ((long)l * HH + h) * EE;
  float s = 0.0f;
#pragma unroll 4
  for (int e = 0; e < EE; ++e) s += xr[e] * wr[e];
  er[idx] = s;
}

// ---------------------------------------------------------------- cell body

// One GRU cell layer-step for this wave's 16x16 tile. B fragments come from
// LDS (weights resident for the whole kernel, bank-conflict-free padded rows);
// A fragments (input row + h row) from global, software-pipelined.
__device__ __forceinline__ void cell_step(
    const unsigned short* aInpRow,     // this lane's input row (k=0)
    const unsigned short* aHRow,       // this lane's h_prev row (k=0)
    const unsigned short* ldsW,        // LDS weight base for this layer
    int laneHi, int laneLo, int mb, int nb,
    const float* __restrict__ bir_p,
    const float* __restrict__ biz_p,
    const float* __restrict__ bin_p,
    const float* __restrict__ er_l,    // [B,H] for this layer
    const float* __restrict__ hprev_f,
    float* __restrict__ hnew_f,
    unsigned short* __restrict__ hnew_bf,
    float* __restrict__ outs,          // null, else [B,T,H]
    int t,
    float* __restrict__ hfinal)        // null, else [B,H]
{
  f32x8 accR = {}, accZ = {}, accN = {}, accHN = {};

  const int kaOff  = laneHi * 16;
  const int kbnOff = laneHi * 16;
  const unsigned short* wRow = ldsW + (size_t)laneLo * KPAD;  // row n of gate 0

  // prologue: stage 0
  bf16x16 aI  = load_a_frag(aInpRow + kaOff);
  bf16x16 aH  = load_a_frag(aHRow + kaOff);
  bf16x16 bIR = load_b_frag(wRow + 0 * 16 * KPAD + kbnOff);
  bf16x16 bHR = load_b_frag(wRow + 1 * 16 * KPAD + kbnOff);
  bf16x16 bIZ = load_b_frag(wRow + 2 * 16 * KPAD + kbnOff);
  bf16x16 bHZ = load_b_frag(wRow + 3 * 16 * KPAD + kbnOff);
  bf16x16 bIN = load_b_frag(wRow + 4 * 16 * KPAD + kbnOff);
  bf16x16 bHN = load_b_frag(wRow + 5 * 16 * KPAD + kbnOff);

  for (int kb = 0; kb < KK - 32; kb += 32) {
    const int ka2  = kb + 32 + kaOff;
    const int kbn2 = kb + 32 + kbnOff;

    bf16x16 aI2  = load_a_frag(aInpRow + ka2);
    bf16x16 aH2  = load_a_frag(aHRow + ka2);
    bf16x16 bIR2 = load_b_frag(wRow + 0 * 16 * KPAD + kbn2);
    bf16x16 bHR2 = load_b_frag(wRow + 1 * 16 * KPAD + kbn2);
    bf16x16 bIZ2 = load_b_frag(wRow + 2 * 16 * KPAD + kbn2);
    bf16x16 bHZ2 = load_b_frag(wRow + 3 * 16 * KPAD + kbn2);
    bf16x16 bIN2 = load_b_frag(wRow + 4 * 16 * KPAD + kbn2);
    bf16x16 bHN2 = load_b_frag(wRow + 5 * 16 * KPAD + kbn2);

    accR  = wmma_bf16(aI, bIR, accR);
    accR  = wmma_bf16(aH, bHR, accR);
    accZ  = wmma_bf16(aI, bIZ, accZ);
    accZ  = wmma_bf16(aH, bHZ, accZ);
    accN  = wmma_bf16(aI, bIN, accN);
    accHN = wmma_bf16(aH, bHN, accHN);

    aI = aI2;  aH = aH2;
    bIR = bIR2; bHR = bHR2;
    bIZ = bIZ2; bHZ = bHZ2;
    bIN = bIN2; bHN = bHN2;
  }

  accR  = wmma_bf16(aI, bIR, accR);
  accR  = wmma_bf16(aH, bHR, accR);
  accZ  = wmma_bf16(aI, bIZ, accZ);
  accZ  = wmma_bf16(aH, bHZ, accZ);
  accN  = wmma_bf16(aI, bIN, accN);
  accHN = wmma_bf16(aH, bHN, accHN);

  // gates: C/D layout -> element r is (m = r + 8*laneHi, n = lane&15)
  const int n_idx = nb + laneLo;
  const float bir = bir_p[n_idx];
  const float biz = biz_p[n_idx];
  const float bin = bin_p[n_idx];

#pragma unroll
  for (int r = 0; r < 8; ++r) {
    const int b_idx = mb + r + laneHi * 8;
    const long hoff = (long)b_idx * HH + n_idx;
    const float rv = sigmoidf_(accR[r] + bir + er_l[hoff]);
    const float zv = sigmoidf_(accZ[r] + biz);
    const float nv = tanhf(accN[r] + bin + rv * accHN[r]);
    const float hp = hprev_f[hoff];
    const float hn = (1.0f - zv) * nv + zv * hp;
    hnew_f[hoff]  = hn;
    hnew_bf[hoff] = f2bf(hn);
    if (outs)   outs[((long)b_idx * TT + t) * HH + n_idx] = hn;
    if (hfinal) hfinal[hoff] = hn;
  }
}

// ---------------------------------------------------------------- persistent kernel

__global__ __launch_bounds__(256) void gru_persistent(
    const unsigned short* __restrict__ x_bf,   // [B,T,I] bf16
    const unsigned short* __restrict__ wbf,    // [12][H][K] bf16 (l*6+gate)
    const float* __restrict__ b_irF,           // [L,H]
    const float* __restrict__ b_izF,
    const float* __restrict__ b_inF,
    const float* __restrict__ er_f,            // [L,B,H]
    float* __restrict__ h_f,                   // [L,2,B,H]
    unsigned short* __restrict__ h_bf,         // [L,2,B,H]
    float* __restrict__ outs,                  // [B,T,H]
    float* __restrict__ hfin,                  // [L,B,H]
    unsigned int* __restrict__ bar)            // [0]=counter, [1]=phase (zeroed)
{
  extern __shared__ unsigned short lds_w[];    // LDS_W_ELEMS halves (padded rows)

  const int lane   = threadIdx.x & 31;
  const int wave   = threadIdx.x >> 5;
  const int laneHi = lane >> 4;
  const int laneLo = lane & 15;
  const int nb     = blockIdx.x << 4;          // this block's hidden slice
  const int mb     = wave << 4;                // this wave's batch slice

  // ---- preload both layers' weight column-slices into LDS via the CDNA5
  //      async-to-LDS datapath. 192 rows (= 12 mats x 16 n) x 512 k halves,
  //      destination rows padded to KPAD halves for bank-conflict-free reads.
  for (int it = 0; it < (LDS_ROWS * (KK / 8)) / 256; ++it) {   // 48 iterations
    const int chunk = threadIdx.x + it * 256;                  // 8-half chunks
    const int k8    = (chunk & 63) << 3;                       // 0..504
    const int row   = chunk >> 6;                              // 0..191 = mat*16+n
    const int mat   = row >> 4;
    const int n     = row & 15;
    const unsigned short* src = wbf + ((size_t)mat * HH + (nb + n)) * KK + k8;
    unsigned      ldsOff = (unsigned)(size_t)(lds_w + (size_t)row * KPAD + k8);
    unsigned long long ga = (unsigned long long)(size_t)src;
    asm volatile("global_load_async_to_lds_b128 %0, %1, off"
                 :: "v"(ldsOff), "v"(ga) : "memory");
  }
  asm volatile("s_wait_asynccnt 0x0" ::: "memory");
  __syncthreads();

  unsigned int phase = 0;   // only threadIdx.x==0 uses it

  for (int t = 0; t < TT; ++t) {
    const int cur = t & 1, nxt = (t + 1) & 1;
    const bool last = (t == TT - 1);

    // ---------------- layer 0: input = x[:, t, :]
    {
      const unsigned short* aInpRow =
          x_bf + ((size_t)(mb + laneLo) * TT + t) * II;
      const unsigned short* aHRow =
          h_bf + ((size_t)(0 * 2 + cur) * BB + (mb + laneLo)) * HH;
      cell_step(aInpRow, aHRow,
                lds_w + (size_t)0 * 6 * 16 * KPAD,
                laneHi, laneLo, mb, nb,
                b_irF + 0 * HH, b_izF + 0 * HH, b_inF + 0 * HH,
                er_f + (size_t)0 * BB * HH,
                h_f  + (size_t)(0 * 2 + cur) * BB * HH,
                h_f  + (size_t)(0 * 2 + nxt) * BB * HH,
                h_bf + (size_t)(0 * 2 + nxt) * BB * HH,
                nullptr, t,
                last ? (hfin + 0 * BB * HH) : nullptr);
    }

    // grid barrier: layer-0 h visible to all blocks
    __syncthreads();
    if (threadIdx.x == 0) {
      __threadfence();
      unsigned arrived = __hip_atomic_fetch_add(&bar[0], 1u, __ATOMIC_ACQ_REL,
                                                __HIP_MEMORY_SCOPE_AGENT);
      if (arrived == NBLK - 1) {
        __hip_atomic_store(&bar[0], 0u, __ATOMIC_RELAXED, __HIP_MEMORY_SCOPE_AGENT);
        __hip_atomic_fetch_add(&bar[1], 1u, __ATOMIC_RELEASE, __HIP_MEMORY_SCOPE_AGENT);
      } else {
        while (__hip_atomic_load(&bar[1], __ATOMIC_ACQUIRE,
                                 __HIP_MEMORY_SCOPE_AGENT) <= phase)
          __builtin_amdgcn_s_sleep(2);
      }
      phase++;
    }
    __syncthreads();

    // ---------------- layer 1: input = layer-0's fresh hidden state
    {
      const unsigned short* aInpRow =
          h_bf + ((size_t)(0 * 2 + nxt) * BB + (mb + laneLo)) * HH;
      const unsigned short* aHRow =
          h_bf + ((size_t)(1 * 2 + cur) * BB + (mb + laneLo)) * HH;
      cell_step(aInpRow, aHRow,
                lds_w + (size_t)1 * 6 * 16 * KPAD,
                laneHi, laneLo, mb, nb,
                b_irF + 1 * HH, b_izF + 1 * HH, b_inF + 1 * HH,
                er_f + (size_t)1 * BB * HH,
                h_f  + (size_t)(1 * 2 + cur) * BB * HH,
                h_f  + (size_t)(1 * 2 + nxt) * BB * HH,
                h_bf + (size_t)(1 * 2 + nxt) * BB * HH,
                outs, t,
                last ? (hfin + 1 * BB * HH) : nullptr);
    }

    // grid barrier: layer-1 h visible before next timestep
    __syncthreads();
    if (threadIdx.x == 0) {
      __threadfence();
      unsigned arrived = __hip_atomic_fetch_add(&bar[0], 1u, __ATOMIC_ACQ_REL,
                                                __HIP_MEMORY_SCOPE_AGENT);
      if (arrived == NBLK - 1) {
        __hip_atomic_store(&bar[0], 0u, __ATOMIC_RELAXED, __HIP_MEMORY_SCOPE_AGENT);
        __hip_atomic_fetch_add(&bar[1], 1u, __ATOMIC_RELEASE, __HIP_MEMORY_SCOPE_AGENT);
      } else {
        while (__hip_atomic_load(&bar[1], __ATOMIC_ACQUIRE,
                                 __HIP_MEMORY_SCOPE_AGENT) <= phase)
          __builtin_amdgcn_s_sleep(2);
      }
      phase++;
    }
    __syncthreads();
  }
}

// ---------------------------------------------------------------- launcher

extern "C" void kernel_launch(void* const* d_in, const int* in_sizes, int n_in,
                              void* d_out, int out_size, void* d_ws, size_t ws_size,
                              hipStream_t stream) {
  const float* x      = (const float*)d_in[0];   // [B,T,I]
  const float* extra  = (const float*)d_in[1];   // [B,E]
  const float* W_irF  = (const float*)d_in[2];   // [L,H,I]
  const float* b_irF  = (const float*)d_in[3];   // [L,H]
  const float* W_hrF  = (const float*)d_in[4];
  const float* W_izF  = (const float*)d_in[5];
  const float* b_izF  = (const float*)d_in[6];
  const float* W_hzF  = (const float*)d_in[7];
  const float* W_inF  = (const float*)d_in[8];
  const float* b_inF  = (const float*)d_in[9];
  const float* W_hnF  = (const float*)d_in[10];
  const float* W_erF  = (const float*)d_in[11];  // [L,H,E]

  // ---- workspace carve-up (~25 MB total)
  char* ws = (char*)d_ws;
  size_t off = 0;
  unsigned short* x_bf = (unsigned short*)(ws + off); off += (size_t)BB * TT * II * 2;
  unsigned short* wbf  = (unsigned short*)(ws + off); off += (size_t)LL * 6 * HH * KK * 2;
  float*          er_f = (float*)(ws + off);          off += (size_t)LL * BB * HH * 4;
  float*          h_f  = (float*)(ws + off);          off += (size_t)LL * 2 * BB * HH * 4;
  unsigned short* h_bf = (unsigned short*)(ws + off); off += (size_t)LL * 2 * BB * HH * 2;
  unsigned int*   bar  = (unsigned int*)(ws + off);   off += 256;

  // ---- zero recurrent state and barrier state (deterministic per call)
  hipMemsetAsync(h_f, 0, (size_t)LL * 2 * BB * HH * 4, stream);
  hipMemsetAsync(h_bf, 0, (size_t)LL * 2 * BB * HH * 2, stream);
  hipMemsetAsync(bar, 0, 256, stream);

  // ---- one-time conversions / precompute
  cvt_f32_to_bf16<<<1024, 256, 0, stream>>>(x, x_bf, BB * TT * II);
  const float* Wsrc[6] = {W_irF, W_hrF, W_izF, W_hzF, W_inF, W_hnF};
  for (int l = 0; l < LL; ++l)
    for (int m = 0; m < 6; ++m)
      cvt_f32_to_bf16<<<256, 256, 0, stream>>>(
          Wsrc[m] + (size_t)l * HH * KK,
          wbf + ((size_t)l * 6 + m) * HH * KK, HH * KK);
  er_kernel<<<(LL * BB * HH + 255) / 256, 256, 0, stream>>>(extra, W_erF, er_f);

  float* outs = (float*)d_out;                       // [B,T,H]
  float* hfin = outs + (size_t)BB * TT * HH;         // [L,B,H]

  // ---- persistent recurrence: 1 launch, LDS-resident weights, grid barriers
  static bool attr_set = false;
  if (!attr_set) {
    hipFuncSetAttribute((const void*)gru_persistent,
                        hipFuncAttributeMaxDynamicSharedMemorySize, LDS_BYTES);
    attr_set = true;
  }
  gru_persistent<<<NBLK, 256, LDS_BYTES, stream>>>(
      x_bf, wbf, b_irF, b_izF, b_inF, er_f, h_f, h_bf, outs, hfin, bar);
}
